// SAGE_74345883894183
// MI455X (gfx1250) — matrix operand order repaired and checked by reading, not compiled, not run on hardware
//
#include <hip/hip_runtime.h>

// ---------------------------------------------------------------------------
// GraphSAGE (2x SAGEConv + linear classifier) for MI455X / gfx1250.
//
// Edge phase: L2-resident gather + float atomics (192MB L2 holds everything).
// Dense phase: v_wmma_f32_16x16x32_bf16 with bf16 hi/lo split ("bf16x3") for
// ~fp32 accuracy at bf16 WMMA rates. One wave32 per 16-row tile.
// ---------------------------------------------------------------------------

typedef __attribute__((ext_vector_type(16))) __bf16 v16bf;
typedef __attribute__((ext_vector_type(8)))  float  v8f;

#define NF 128  // feature width for all hidden layers

__device__ __forceinline__ v8f wmma_bf16(v16bf a, v16bf b, v8f c) {
    // D = A*B + C, f32 accumulate
    return __builtin_amdgcn_wmma_f32_16x16x32_bf16(
        false, a, false, b, (short)0, c, false, false);
}

// Load the 16-element A fragment for one lane (one contiguous run of 16 fp32),
// scale, and split into bf16 hi/lo.
__device__ __forceinline__ void load_split_frag(const float* __restrict__ base,
                                                float scale, v16bf& hi, v16bf& lo) {
    float4 r0 = *(const float4*)(base + 0);
    float4 r1 = *(const float4*)(base + 4);
    float4 r2 = *(const float4*)(base + 8);
    float4 r3 = *(const float4*)(base + 12);
    float v[16] = {r0.x, r0.y, r0.z, r0.w, r1.x, r1.y, r1.z, r1.w,
                   r2.x, r2.y, r2.z, r2.w, r3.x, r3.y, r3.z, r3.w};
#pragma unroll
    for (int j = 0; j < 16; ++j) {
        float f = v[j] * scale;
        __bf16 h = (__bf16)f;
        hi[j] = h;
        lo[j] = (__bf16)(f - (float)h);
    }
}

// ---------------------------- small prep kernels ---------------------------

__global__ void split_weight(const float* __restrict__ src,
                             __bf16* __restrict__ hi, __bf16* __restrict__ lo, int n) {
    int i = blockIdx.x * blockDim.x + threadIdx.x;
    if (i >= n) return;
    float v = src[i];
    __bf16 h = (__bf16)v;
    hi[i] = h;
    lo[i] = (__bf16)(v - (float)h);
}

// Pad classifier weights 40x128 -> 48x128 with zero rows, split hi/lo.
__global__ void split_weight_pad(const float* __restrict__ src,
                                 __bf16* __restrict__ hi, __bf16* __restrict__ lo,
                                 int rows_src, int rows_pad, int cols) {
    int i = blockIdx.x * blockDim.x + threadIdx.x;
    if (i >= rows_pad * cols) return;
    int r = i / cols, c = i % cols;
    float v = (r < rows_src) ? src[r * cols + c] : 0.0f;
    __bf16 h = (__bf16)v;
    hi[i] = h;
    lo[i] = (__bf16)(v - (float)h);
}

__global__ void degree_count(const int* __restrict__ dst, float* __restrict__ deg, int nE) {
    int e = blockIdx.x * blockDim.x + threadIdx.x;
    if (e < nE) atomicAdd(&deg[dst[e]], 1.0f);
}

__global__ void invert_deg(float* __restrict__ deg, int n) {
    int i = blockIdx.x * blockDim.x + threadIdx.x;
    if (i < n) deg[i] = 1.0f / fmaxf(deg[i], 1.0f);
}

// 32 lanes per edge, one float4 per lane (128 floats per edge). L2-resident.
__global__ void scatter_accum(const float* __restrict__ feat,
                              const int* __restrict__ src, const int* __restrict__ dst,
                              float* __restrict__ aggr, int nE) {
    long long tid = (long long)blockIdx.x * blockDim.x + threadIdx.x;
    int e = (int)(tid >> 5);
    int l = (int)(tid & 31);
    if (e >= nE) return;
    int s = src[e], d = dst[e];
    float4 v = ((const float4*)(feat + (size_t)s * NF))[l];
    float* o = aggr + (size_t)d * NF + l * 4;
    atomicAdd(o + 0, v.x);
    atomicAdd(o + 1, v.y);
    atomicAdd(o + 2, v.z);
    atomicAdd(o + 3, v.w);
}

// --------------------------- SAGE fused GEMM -------------------------------
// out[16t..16t+15, :] = act( deg_inv*aggr @ Wl^T + b + X @ Wr^T )
// One wave per 16-row tile; 8 N-tiles of 16 cols; K in 4 chunks of 32.
__global__ __launch_bounds__(256)
void sage_gemm(const float* __restrict__ Aag, const float* __restrict__ Ax,
               const __bf16* __restrict__ Wlh, const __bf16* __restrict__ Wll,
               const __bf16* __restrict__ Wrh, const __bf16* __restrict__ Wrl,
               const float* __restrict__ bias, const float* __restrict__ dinv,
               float* __restrict__ out, int ntiles, int do_relu) {
    int tile = blockIdx.x * blockDim.y + threadIdx.y;   // wave-uniform
    if (tile >= ntiles) return;                          // EXEC stays all-ones
    int lane = threadIdx.x;
    int half = lane >> 4, mrow = lane & 15;
    int arow = tile * 16 + mrow;                         // this lane's A row
    float s = dinv[arow];
    const float* ag = Aag + (size_t)arow * NF + half * 16;
    const float* ax = Ax  + (size_t)arow * NF + half * 16;

    for (int n = 0; n < 8; ++n) {
        v8f acc = {};
        size_t brow = (size_t)(n * 16 + mrow) * NF + half * 16;
#pragma unroll
        for (int k = 0; k < 4; ++k) {
            int kb = k * 32;
            v16bf a1h, a1l, a2h, a2l;
            load_split_frag(ag + kb, s,    a1h, a1l);    // mean-div fused here
            load_split_frag(ax + kb, 1.0f, a2h, a2l);
            v16bf b1h = *(const v16bf*)(Wlh + brow + kb);
            v16bf b1l = *(const v16bf*)(Wll + brow + kb);
            v16bf b2h = *(const v16bf*)(Wrh + brow + kb);
            v16bf b2l = *(const v16bf*)(Wrl + brow + kb);
            // bf16x3: hi*hi + hi*lo + lo*hi  ~ fp32 product
            acc = wmma_bf16(a1h, b1h, acc);
            acc = wmma_bf16(a1h, b1l, acc);
            acc = wmma_bf16(a1l, b1h, acc);
            acc = wmma_bf16(a2h, b2h, acc);
            acc = wmma_bf16(a2h, b2l, acc);
            acc = wmma_bf16(a2l, b2h, acc);
        }
        int col = n * 16 + mrow;
        float bv = bias[col];
        int rbase = tile * 16 + half * 8;
#pragma unroll
        for (int r = 0; r < 8; ++r) {
            float o = acc[r] + bv;
            if (do_relu) o = fmaxf(o, 0.0f);
            out[(size_t)(rbase + r) * NF + col] = o;
        }
    }
}

// --------------------------- classifier GEMM -------------------------------
// out[N,40] = h1 @ Wcls^T + b ; Wcls padded to 48 rows (3 N-tiles).
__global__ __launch_bounds__(256)
void cls_gemm(const float* __restrict__ A,
              const __bf16* __restrict__ Wh, const __bf16* __restrict__ Wl,
              const float* __restrict__ bias, float* __restrict__ out,
              int ntiles, int ncls) {
    int tile = blockIdx.x * blockDim.y + threadIdx.y;
    if (tile >= ntiles) return;
    int lane = threadIdx.x;
    int half = lane >> 4, mrow = lane & 15;
    int arow = tile * 16 + mrow;
    const float* ap = A + (size_t)arow * NF + half * 16;

    for (int n = 0; n < 3; ++n) {
        v8f acc = {};
        size_t brow = (size_t)(n * 16 + mrow) * NF + half * 16;
#pragma unroll
        for (int k = 0; k < 4; ++k) {
            int kb = k * 32;
            v16bf ah, al;
            load_split_frag(ap + kb, 1.0f, ah, al);
            v16bf bh = *(const v16bf*)(Wh + brow + kb);
            v16bf bl = *(const v16bf*)(Wl + brow + kb);
            acc = wmma_bf16(ah, bh, acc);
            acc = wmma_bf16(ah, bl, acc);
            acc = wmma_bf16(al, bh, acc);
        }
        int col = n * 16 + mrow;
        if (col < ncls) {
            float bv = bias[col];
            int rbase = tile * 16 + half * 8;
#pragma unroll
            for (int r = 0; r < 8; ++r)
                out[(size_t)(rbase + r) * ncls + col] = acc[r] + bv;
        }
    }
}

// ------------------------------- launcher ----------------------------------

extern "C" void kernel_launch(void* const* d_in, const int* in_sizes, int n_in,
                              void* d_out, int out_size, void* d_ws, size_t ws_size,
                              hipStream_t stream) {
    const float* x    = (const float*)d_in[0];
    const int*   ei   = (const int*)d_in[1];     // [2,E] row-major
    const float* Wl0  = (const float*)d_in[2];
    const float* bl0  = (const float*)d_in[3];
    const float* Wr0  = (const float*)d_in[4];
    const float* Wl1  = (const float*)d_in[5];
    const float* bl1  = (const float*)d_in[6];
    const float* Wr1  = (const float*)d_in[7];
    const float* Wcls = (const float*)d_in[8];
    const float* bcls = (const float*)d_in[9];
    float* out = (float*)d_out;

    const int nNodes = in_sizes[0] / NF;      // 100000 (multiple of 16)
    const int nE     = in_sizes[1] / 2;       // 1600000
    const int ncls   = in_sizes[9];           // 40
    const int nclsPad = 48;
    const int ntiles = nNodes / 16;           // 6250
    const int* src = ei;
    const int* dst = ei + nE;

    // ---- carve workspace -------------------------------------------------
    char* ws = (char*)d_ws;
    size_t off = 0;
    auto carve = [&](size_t bytes) -> void* {
        void* p = ws + off;
        off += (bytes + 255) & ~(size_t)255;
        return p;
    };
    float*  dinv  = (float*)carve((size_t)nNodes * 4);
    float*  aggr  = (float*)carve((size_t)nNodes * NF * 4);
    float*  h0    = (float*)carve((size_t)nNodes * NF * 4);
    float*  h1    = (float*)carve((size_t)nNodes * NF * 4);
    __bf16* Wl0h = (__bf16*)carve(NF * NF * 2);
    __bf16* Wl0l = (__bf16*)carve(NF * NF * 2);
    __bf16* Wr0h = (__bf16*)carve(NF * NF * 2);
    __bf16* Wr0l = (__bf16*)carve(NF * NF * 2);
    __bf16* Wl1h = (__bf16*)carve(NF * NF * 2);
    __bf16* Wl1l = (__bf16*)carve(NF * NF * 2);
    __bf16* Wr1h = (__bf16*)carve(NF * NF * 2);
    __bf16* Wr1l = (__bf16*)carve(NF * NF * 2);
    __bf16* Wch  = (__bf16*)carve(nclsPad * NF * 2);
    __bf16* Wcl  = (__bf16*)carve(nclsPad * NF * 2);
    (void)ws_size; (void)n_in; (void)out_size;

    // ---- degree / deg_inv ------------------------------------------------
    hipMemsetAsync(dinv, 0, (size_t)nNodes * 4, stream);
    degree_count<<<(nE + 255) / 256, 256, 0, stream>>>(dst, dinv, nE);
    invert_deg<<<(nNodes + 255) / 256, 256, 0, stream>>>(dinv, nNodes);

    // ---- weight prep (bf16 hi/lo split) ----------------------------------
    const int wn = NF * NF;
    split_weight<<<(wn + 255) / 256, 256, 0, stream>>>(Wl0, Wl0h, Wl0l, wn);
    split_weight<<<(wn + 255) / 256, 256, 0, stream>>>(Wr0, Wr0h, Wr0l, wn);
    split_weight<<<(wn + 255) / 256, 256, 0, stream>>>(Wl1, Wl1h, Wl1l, wn);
    split_weight<<<(wn + 255) / 256, 256, 0, stream>>>(Wr1, Wr1h, Wr1l, wn);
    split_weight_pad<<<(nclsPad * NF + 255) / 256, 256, 0, stream>>>(
        Wcls, Wch, Wcl, ncls, nclsPad, NF);

    dim3 gblk(32, 8);                                   // 8 waves per block
    int ggrd = (ntiles + 7) / 8;
    long long sthreads = (long long)nE * 32;
    int sgrd = (int)((sthreads + 255) / 256);

    // ---- layer 0 ---------------------------------------------------------
    hipMemsetAsync(aggr, 0, (size_t)nNodes * NF * 4, stream);
    scatter_accum<<<sgrd, 256, 0, stream>>>(x, src, dst, aggr, nE);
    sage_gemm<<<ggrd, gblk, 0, stream>>>(aggr, x, Wl0h, Wl0l, Wr0h, Wr0l,
                                         bl0, dinv, h0, ntiles, 1);

    // ---- layer 1 ---------------------------------------------------------
    hipMemsetAsync(aggr, 0, (size_t)nNodes * NF * 4, stream);
    scatter_accum<<<sgrd, 256, 0, stream>>>(h0, src, dst, aggr, nE);
    sage_gemm<<<ggrd, gblk, 0, stream>>>(aggr, h0, Wl1h, Wl1l, Wr1h, Wr1l,
                                         bl1, dinv, h1, ntiles, 1);

    // ---- classifier ------------------------------------------------------
    cls_gemm<<<ggrd, gblk, 0, stream>>>(h1, Wch, Wcl, bcls, out, ntiles, ncls);
}
